// GCN_36309653521103
// MI455X (gfx1250) — matrix operand order repaired and checked by reading, not compiled, not run on hardware
//
#include <hip/hip_runtime.h>

typedef __attribute__((ext_vector_type(16))) __bf16 v16bf;
typedef __attribute__((ext_vector_type(8)))  float  v8f;

// ---------------------------------------------------------------- utilities
__global__ void k_zero(float* __restrict__ p, long n) {
  long i = (long)blockIdx.x * blockDim.x + threadIdx.x;
  long stride = (long)gridDim.x * blockDim.x;
  for (; i < n; i += stride) p[i] = 0.0f;
}

// deg[c] += 1 over all edges (col side) + self loops
__global__ void k_degree(const int* __restrict__ col, int E, int Nn,
                         float* __restrict__ deg) {
  int total = E + Nn;
  int stride = gridDim.x * blockDim.x;
  for (int i = blockIdx.x * blockDim.x + threadIdx.x; i < total; i += stride) {
    int c = (i < E) ? col[i] : (i - E);
    atomicAdd(&deg[c], 1.0f);
  }
}

// deg -> d^{-1/2} in place
__global__ void k_dinv(float* __restrict__ deg, int Nn) {
  int i = blockIdx.x * blockDim.x + threadIdx.x;
  if (i < Nn) {
    float d = deg[i];
    deg[i] = (d > 0.0f) ? rsqrtf(d) : 0.0f;
  }
}

// W [Kact x Nact] row-major f32  ->  Wt [Npad x Kpad] row-major bf16, zero padded
__global__ void k_prep_w(const float* __restrict__ W, __bf16* __restrict__ Wt,
                         int Kact, int Nact, int Kpad, int Npad) {
  int idx = blockIdx.x * blockDim.x + threadIdx.x;
  if (idx < Kpad * Npad) {
    int n = idx / Kpad, k = idx % Kpad;
    float v = (k < Kact && n < Nact) ? W[k * Nact + n] : 0.0f;
    Wt[idx] = (__bf16)v;
  }
}

// ---------------------------------------------------------------- WMMA GEMM
// Each wave computes a 16-row M tile x all NT 16-col N tiles.
// A: f32 row-major [M x INSTRIDE], valid K dims = KACT (rest read as 0)
// Wt: bf16 [NT*16 x KPAD] row-major (pre-transposed weights)
// Hout: f32 row-major [M x OUTSTRIDE]
template <int NT, int KCH, int KACT, int KPAD, int INSTRIDE, int OUTSTRIDE>
__global__ void k_gemm_wmma(const float* __restrict__ A,
                            const __bf16* __restrict__ Wt,
                            float* __restrict__ Hout, int Mtiles) {
  int wave = blockIdx.x * (blockDim.x >> 5) + (threadIdx.x >> 5);
  if (wave >= Mtiles) return;          // wave-uniform: EXEC stays all-ones
  int lane = threadIdx.x & 31;
  int mlo  = lane & 15;
  int hi   = lane >> 4;                // half-wave select
  int m0   = wave * 16;

  v8f acc[NT] = {};

  const float* arow = A + (long)(m0 + mlo) * INSTRIDE;

  for (int kc = 0; kc < KCH; ++kc) {
    const int kbase = kc * 32;
    // --- A fragment: 16-bit A 16x32, contiguous k order (same as B) ---
    // lanes 0-15: M=0..15, K=0..15 contiguous
    // lanes 16-31: same M, K=16..31 contiguous
    v16bf a;
#pragma unroll
    for (int e = 0; e < 16; ++e) {
      int g = e >> 1, p = e & 1;
      int k = kbase + hi * 16 + 2 * g + p;
      float v = (k < KACT) ? arow[k] : 0.0f;
      a[e] = (__bf16)v;
    }
    // --- B fragments + WMMA per N tile ---
    // B 32x16: lane n = lane&15; lanes 0-15 hold K=kbase..kbase+15,
    // lanes 16-31 hold K=kbase+16..kbase+31 -> 16 contiguous bf16 per lane.
#pragma unroll
    for (int t = 0; t < NT; ++t) {
      const __bf16* bp = Wt + (long)(t * 16 + mlo) * KPAD + kbase + hi * 16;
      v16bf b = *(const v16bf*)bp;
      acc[t] = __builtin_amdgcn_wmma_f32_16x16x32_bf16(
          false, a, false, b, (short)0, acc[t], false, false);
    }
  }

  // C/D layout: VGPR r -> M = r + hi*8, N = lane&15
#pragma unroll
  for (int t = 0; t < NT; ++t) {
#pragma unroll
    for (int r = 0; r < 8; ++r) {
      int m = m0 + r + hi * 8;
      int n = t * 16 + mlo;
      Hout[(long)m * OUTSTRIDE + n] = acc[t][r];
    }
  }
}

// ---------------------------------------------------------------- scatter
// One wave per edge: out[col] += h[row] * dinv[row]*dinv[col]
template <int DIM, int STRIDE>
__global__ void k_scatter(const int* __restrict__ row, const int* __restrict__ col,
                          int E, int Nn, const float* __restrict__ dinv,
                          const float* __restrict__ H, float* __restrict__ Agg) {
  int wid  = blockIdx.x * (blockDim.x >> 5) + (threadIdx.x >> 5);
  int lane = threadIdx.x & 31;
  if (wid >= E + Nn) return;
  int r, c;
  if (wid < E) { r = row[wid]; c = col[wid]; }
  else         { r = wid - E;  c = r; }
  float nrm = dinv[r] * dinv[c];
  const float* hr = H + (long)r * STRIDE;
  float* ac = Agg + (long)c * STRIDE;
  for (int d = lane; d < DIM; d += 32)
    atomicAdd(&ac[d], hr[d] * nrm);
}

// layer-1 bias + relu in place on padded [N x 80] buffer (cols 0..64)
__global__ void k_bias_relu(float* __restrict__ Agg, const float* __restrict__ b,
                            int Nn) {
  long i = (long)blockIdx.x * blockDim.x + threadIdx.x;
  long tot = (long)Nn * 65;
  if (i < tot) {
    int m = (int)(i / 65), d = (int)(i % 65);
    float v = Agg[(long)m * 80 + d] + b[d];
    Agg[(long)m * 80 + d] = fmaxf(v, 0.0f);
  }
}

// fused: relu(Agg2 + b2) . Wout + bout, relu -> out   (one wave per node)
__global__ void k_final(const float* __restrict__ Agg2, const float* __restrict__ b2,
                        const float* __restrict__ Wout, const float* __restrict__ bout,
                        float* __restrict__ out, int Nn) {
  int wid  = blockIdx.x * (blockDim.x >> 5) + (threadIdx.x >> 5);
  int lane = threadIdx.x & 31;
  if (wid >= Nn) return;
  const float* a = Agg2 + (long)wid * 64;
  float acc = 0.0f;
#pragma unroll
  for (int s = 0; s < 2; ++s) {
    int d = lane + s * 32;
    float h = fmaxf(a[d] + b2[d], 0.0f);
    acc += h * Wout[d];
  }
#pragma unroll
  for (int off = 16; off > 0; off >>= 1)
    acc += __shfl_xor(acc, off, 32);
  if (lane == 0) out[wid] = fmaxf(acc + bout[0], 0.0f);
}

// ---------------------------------------------------------------- launcher
extern "C" void kernel_launch(void* const* d_in, const int* in_sizes, int n_in,
                              void* d_out, int out_size, void* d_ws, size_t ws_size,
                              hipStream_t stream) {
  const float* x    = (const float*)d_in[0];
  const int*   ei   = (const int*)d_in[1];
  const float* W1   = (const float*)d_in[2];
  const float* b1   = (const float*)d_in[3];
  const float* W2   = (const float*)d_in[4];
  const float* b2   = (const float*)d_in[5];
  const float* Wout = (const float*)d_in[6];
  const float* bout = (const float*)d_in[7];
  float* out = (float*)d_out;

  const int Nn = in_sizes[0] / 100;   // 200000
  const int E  = in_sizes[1] / 2;     // 6400000
  const int* row = ei;
  const int* col = ei + E;

  // workspace carving (256B aligned)
  char* ws = (char*)d_ws;
  size_t off = 0;
  auto carve = [&](size_t bytes) -> void* {
    void* p = ws + off;
    off = (off + bytes + 255) & ~(size_t)255;
    return p;
  };
  float*  dinv = (float*)carve((size_t)Nn * 4);        //  0.8 MB
  float*  H    = (float*)carve((size_t)Nn * 80 * 4);   // 64 MB (reused layer 2)
  float*  A1   = (float*)carve((size_t)Nn * 80 * 4);   // 64 MB
  float*  A2   = (float*)carve((size_t)Nn * 64 * 4);   // 51 MB
  __bf16* Wt1  = (__bf16*)carve(80 * 128 * 2);
  __bf16* Wt2  = (__bf16*)carve(64 * 96 * 2);

  // init
  k_zero<<<2048, 256, 0, stream>>>(dinv, (long)Nn);
  k_zero<<<4096, 256, 0, stream>>>(A1, (long)Nn * 80);
  k_zero<<<4096, 256, 0, stream>>>(A2, (long)Nn * 64);

  // symmetric normalization coefficients
  k_degree<<<4096, 256, 0, stream>>>(col, E, Nn, dinv);
  k_dinv<<<(Nn + 255) / 256, 256, 0, stream>>>(dinv, Nn);

  // weight prep (bf16, transposed, zero-padded)
  k_prep_w<<<(128 * 80 + 255) / 256, 256, 0, stream>>>(W1, Wt1, 100, 65, 128, 80);
  k_prep_w<<<(96 * 64 + 255) / 256, 256, 0, stream>>>(W2, Wt2, 65, 64, 96, 64);

  const int Mtiles = Nn / 16;                 // 12500 (Nn % 16 == 0)
  const int gemmBlocks = (Mtiles + 3) / 4;    // 4 waves / 128-thread block
  const int wgrid = (E + Nn + 7) / 8;         // 8 edge-waves / 256-thread block

  // layer 1: H = X @ W1 (bf16 WMMA, f32 acc), then normalized scatter-add
  k_gemm_wmma<5, 4, 100, 128, 100, 80><<<gemmBlocks, 128, 0, stream>>>(x, Wt1, H, Mtiles);
  k_scatter<65, 80><<<wgrid, 256, 0, stream>>>(row, col, E, Nn, dinv, H, A1);
  k_bias_relu<<<(int)(((long)Nn * 65 + 255) / 256), 256, 0, stream>>>(A1, b1, Nn);

  // layer 2
  k_gemm_wmma<4, 3, 65, 96, 80, 64><<<gemmBlocks, 128, 0, stream>>>(A1, Wt2, H, Mtiles);
  k_scatter<64, 64><<<wgrid, 256, 0, stream>>>(row, col, E, Nn, dinv, H, A2);

  // readout
  k_final<<<(Nn + 7) / 8, 256, 0, stream>>>(A2, b2, Wout, bout, out, Nn);
}
